// ChaoticDecoder_64149631533056
// MI455X (gfx1250) — matrix operand order repaired and checked by reading, not compiled, not run on hardware
//
#include <hip/hip_runtime.h>
#include <hip/hip_bf16.h>
#include <math.h>

// Problem constants (from reference): BS=32, SEQ=64, H=256, OUT=1
#define HDIM 256
#define SEQL 64
#define BSZ  32

typedef __attribute__((ext_vector_type(16))) _Float16 v16h;
typedef __attribute__((ext_vector_type(8)))  float    v8f;

union AFrag { v16h v; float4 q[2]; };

// ---------- WMMA fragment helpers (wave32, 16x16x32 f16) ----------
// A layout (16x32 f16, MxK): lane L -> M = L%16, g = L/16.
//   halves 0..7  = K = k0 + 8g .. k0+8g+7      (contiguous)
//   halves 8..15 = K = k0 + 16 + 8g .. +7      (contiguous)
__device__ __forceinline__ v16h wmma_load_a(const _Float16* base, int ld, int k0, int lane) {
  int m = lane & 15, g = lane >> 4;
  AFrag f;
  const _Float16* p = base + (size_t)m * ld + k0 + 8 * g;
  f.q[0] = *(const float4*)p;        // halves 0..7
  f.q[1] = *(const float4*)(p + 16); // halves 8..15
  return f.v;
}

// B fragments are pre-packed: block (kt,nt) = 512 contiguous halves,
// lane L owns halves [L*16 .. L*16+15] (two 16B loads).
__device__ __forceinline__ v16h wmma_load_b(const _Float16* pack, int kt, int nt, int ntiles, int lane) {
  AFrag f;
  const _Float16* p = pack + (((size_t)(kt * ntiles + nt)) << 9) + lane * 16;
  f.q[0] = *(const float4*)p;
  f.q[1] = *(const float4*)(p + 8);
  return f.v;
}

__device__ __forceinline__ v8f wmma_f16(v16h a, v16h b, v8f c) {
  return __builtin_amdgcn_wmma_f32_16x16x32_f16(false, a, false, b, (short)0, c, false, false);
}

__device__ __forceinline__ float sigm(float x) { return 1.0f / (1.0f + __expf(-x)); }

// ---------- Kernel: f32 -> f16 elementwise ----------
__global__ void cvt_f16_kernel(const float* __restrict__ src, _Float16* __restrict__ dst, int n) {
  int i = blockIdx.x * blockDim.x + threadIdx.x;
  if (i < n) dst[i] = (_Float16)src[i];
}

// ---------- Kernel: pack row-major KxN f32 weight into WMMA-B f16 blocks ----------
// dst half index = block*512 + lane*16 + h, where
//   n = nt*16 + (lane%16), k = kt*32 + (h<8 ? 8*(lane/16) + h : 16 + 8*(lane/16) + h-8),  block = kt*ntiles + nt
__global__ void pack_b_kernel(const float* __restrict__ W, _Float16* __restrict__ dst, int K, int N) {
  int idx = blockIdx.x * blockDim.x + threadIdx.x;
  int total = K * N;
  if (idx >= total) return;
  int block = idx >> 9;
  int rem   = idx & 511;
  int lane  = rem >> 4;
  int h     = rem & 15;
  int ntiles = N >> 4;
  int kt = block / ntiles, nt = block - kt * ntiles;
  int n = nt * 16 + (lane & 15);
  int k = kt * 32 + 8 * (lane >> 4) + (h & 7) + ((h >> 3) << 4);
  dst[idx] = (_Float16)W[(size_t)k * N + n];
}

// ---------- Kernel: Xa[row, n] = x[row,:] @ Wa_x + ba  (rows = 2048, K = N = 256) ----------
__global__ void xa_gemm_kernel(const _Float16* __restrict__ x_h, const _Float16* __restrict__ waX,
                               const float* __restrict__ ba, float* __restrict__ Xa) {
  int gw   = (blockIdx.x * blockDim.x + threadIdx.x) >> 5; // 2048 waves
  int lane = threadIdx.x & 31;
  int mt = gw >> 4;   // 0..127
  int nt = gw & 15;   // 0..15
  v8f acc = {};
  const _Float16* abase = x_h + (size_t)mt * 16 * HDIM;
#pragma unroll
  for (int kt = 0; kt < 8; ++kt) {
    v16h a = wmma_load_a(abase, HDIM, kt * 32, lane);
    v16h b = wmma_load_b(waX, kt, nt, 16, lane);
    acc = wmma_f16(a, b, acc);
  }
  int n = nt * 16 + (lane & 15);
  int g = lane >> 4;
  float bav = ba[n];
#pragma unroll
  for (int r = 0; r < 8; ++r) {
    int m = r + 8 * g;
    Xa[(size_t)(mt * 16 + m) * HDIM + n] = acc[r] + bav;
  }
}

// ---------- Kernel: persistent sequential scan (2 blocks x 256 threads, 16 batch rows each) ----------
__global__ void scan_kernel(const float* __restrict__ x, const float* __restrict__ Xa,
                            const _Float16* __restrict__ waH, const _Float16* __restrict__ waC,
                            const _Float16* __restrict__ wiP, const _Float16* __restrict__ whP,
                            const float* __restrict__ bvec, const float* __restrict__ Wf,
                            const float* __restrict__ bf, float* __restrict__ out) {
  extern __shared__ char smem_raw[];
  float*    ht    = (float*)smem_raw;            // 16*256 f32
  float*    ct    = ht + 16 * HDIM;              // 16*256 f32
  float*    biasS = ct + 16 * HDIM;              // 16*256 f32
  float*    gates = biasS + 16 * HDIM;           // 16*1024 f32
  _Float16* ht_h  = (_Float16*)(gates + 16 * 1024);
  _Float16* ct_h  = ht_h + 16 * HDIM;
  _Float16* cx_h  = ct_h + 16 * HDIM;

  const int tid  = threadIdx.x;
  const int wave = tid >> 5;
  const int lane = tid & 31;
  const int b0   = blockIdx.x * 16; // global batch offset of this workgroup

  for (int i = tid; i < 16 * HDIM; i += 256) {
    ht[i] = 0.f; ct[i] = 0.f;
    ht_h[i] = (_Float16)0.f; ct_h[i] = (_Float16)0.f;
  }
  __syncthreads();

  for (int step = 0; step < SEQL; ++step) {
    // --- (a) attention bias = ht @ Wa_h + ct @ Wa_c   (16 x 256, K=256 each) ---
#pragma unroll
    for (int t = 0; t < 2; ++t) {
      int nt = wave * 2 + t; // 0..15
      v8f acc = {};
#pragma unroll
      for (int kt = 0; kt < 8; ++kt) {
        v16h a  = wmma_load_a(ht_h, HDIM, kt * 32, lane);
        v16h bb = wmma_load_b(waH, kt, nt, 16, lane);
        acc = wmma_f16(a, bb, acc);
        a  = wmma_load_a(ct_h, HDIM, kt * 32, lane);
        bb = wmma_load_b(waC, kt, nt, 16, lane);
        acc = wmma_f16(a, bb, acc);
      }
      int n = nt * 16 + (lane & 15);
      int g = lane >> 4;
#pragma unroll
      for (int r = 0; r < 8; ++r) biasS[(r + 8 * g) * HDIM + n] = acc[r];
    }
    __syncthreads();

    // --- (b) online softmax over seq + context (per (batch,channel), fully parallel) ---
#pragma unroll 1
    for (int c = 0; c < 16; ++c) {
      int ch = tid + c * 256;          // 0..4095
      int b  = ch >> 8;                // local batch 0..15
      int n  = ch & 255;               // channel
      float bb = biasS[b * HDIM + n];
      const float* xa_p = Xa + ((size_t)(b0 + b) * SEQL) * HDIM + n;
      const float* x_p  = x  + ((size_t)(b0 + b) * SEQL) * HDIM + n;
      float m = -INFINITY, l = 0.f, acc = 0.f;
#pragma unroll 1
      for (int s = 0; s < SEQL; ++s) {
        float sc   = xa_p[s * HDIM] + bb;
        float xm   = fmaxf(m, sc);
        float corr = __expf(m - xm);
        float e    = __expf(sc - xm);
        l   = l * corr + e;
        acc = acc * corr + e * x_p[s * HDIM];
        m   = xm;
      }
      cx_h[b * HDIM + n] = (_Float16)(acc / l);
    }
    __syncthreads();

    // --- (c) gates = context @ Wi + ht @ Wh + b   (16 x 1024, K=256 each) ---
#pragma unroll
    for (int t = 0; t < 8; ++t) {
      int nt = wave * 8 + t; // 0..63
      v8f acc = {};
#pragma unroll
      for (int kt = 0; kt < 8; ++kt) {
        v16h a  = wmma_load_a(cx_h, HDIM, kt * 32, lane);
        v16h bb = wmma_load_b(wiP, kt, nt, 64, lane);
        acc = wmma_f16(a, bb, acc);
        a  = wmma_load_a(ht_h, HDIM, kt * 32, lane);
        bb = wmma_load_b(whP, kt, nt, 64, lane);
        acc = wmma_f16(a, bb, acc);
      }
      int n = nt * 16 + (lane & 15);
      int g = lane >> 4;
      float bg = bvec[n];
#pragma unroll
      for (int r = 0; r < 8; ++r) gates[(r + 8 * g) * 1024 + n] = acc[r] + bg;
    }
    __syncthreads();

    // --- (d) LSTM pointwise: order i,f,g,o ---
#pragma unroll 1
    for (int c = 0; c < 16; ++c) {
      int ch = tid + c * 256;
      int b = ch >> 8, n = ch & 255;
      float gi = gates[b * 1024 + n];
      float gf = gates[b * 1024 + 256 + n];
      float gg = gates[b * 1024 + 512 + n];
      float go = gates[b * 1024 + 768 + n];
      float ctn = sigm(gf) * ct[b * HDIM + n] + sigm(gi) * tanhf(gg);
      float htn = sigm(go) * tanhf(ctn);
      ct[b * HDIM + n] = ctn;
      ht[b * HDIM + n] = htn;
      ct_h[b * HDIM + n] = (_Float16)ctn;
      ht_h[b * HDIM + n] = (_Float16)htn;
    }
    __syncthreads();
  }

  // --- final fc: out[b] = ht[b] @ Wf + bf   (OUT = 1) ---
  if (tid < 16) {
    float s = bf[0];
#pragma unroll 1
    for (int k = 0; k < HDIM; ++k) s += ht[tid * HDIM + k] * Wf[k];
    out[b0 + tid] = s;
  }
}

// ---------------------------------------------------------------------------
extern "C" void kernel_launch(void* const* d_in, const int* in_sizes, int n_in,
                              void* d_out, int out_size, void* d_ws, size_t ws_size,
                              hipStream_t stream) {
  const float* x  = (const float*)d_in[0]; // (32,64,256)
  const float* Wa = (const float*)d_in[1]; // (768,256)
  const float* ba = (const float*)d_in[2]; // (256)
  const float* Wi = (const float*)d_in[3]; // (256,1024)
  const float* Wh = (const float*)d_in[4]; // (256,1024)
  const float* bv = (const float*)d_in[5]; // (1024)
  const float* Wf = (const float*)d_in[6]; // (256,1)
  const float* bf = (const float*)d_in[7]; // (1)
  float* out = (float*)d_out;              // (32,1)

  // workspace carve-up
  char* ws = (char*)d_ws;
  _Float16* x_h  = (_Float16*)ws;                 ws += (size_t)BSZ * SEQL * HDIM * 2; // 1 MB
  _Float16* waX  = (_Float16*)ws;                 ws += (size_t)HDIM * HDIM * 2;       // 128 KB
  _Float16* waH  = (_Float16*)ws;                 ws += (size_t)HDIM * HDIM * 2;
  _Float16* waC  = (_Float16*)ws;                 ws += (size_t)HDIM * HDIM * 2;
  _Float16* wiP  = (_Float16*)ws;                 ws += (size_t)HDIM * 4 * HDIM * 2;   // 512 KB
  _Float16* whP  = (_Float16*)ws;                 ws += (size_t)HDIM * 4 * HDIM * 2;
  float*    Xa   = (float*)ws;                    // 2 MB

  const int nx = BSZ * SEQL * HDIM; // 524288
  cvt_f16_kernel<<<(nx + 255) / 256, 256, 0, stream>>>(x, x_h, nx);

  // pack the three 256x256 slices of Wa (rows 0..255 / 256..511 / 512..767)
  pack_b_kernel<<<(HDIM * HDIM + 255) / 256, 256, 0, stream>>>(Wa,                waX, HDIM, HDIM);
  pack_b_kernel<<<(HDIM * HDIM + 255) / 256, 256, 0, stream>>>(Wa + HDIM * HDIM,     waH, HDIM, HDIM);
  pack_b_kernel<<<(HDIM * HDIM + 255) / 256, 256, 0, stream>>>(Wa + 2 * HDIM * HDIM, waC, HDIM, HDIM);
  pack_b_kernel<<<(HDIM * 4 * HDIM + 255) / 256, 256, 0, stream>>>(Wi, wiP, HDIM, 4 * HDIM);
  pack_b_kernel<<<(HDIM * 4 * HDIM + 255) / 256, 256, 0, stream>>>(Wh, whP, HDIM, 4 * HDIM);

  // Xa = x @ Wa_x + ba : 2048 wave-tiles -> 256 blocks of 256 threads (8 waves)
  xa_gemm_kernel<<<256, 256, 0, stream>>>(x_h, waX, ba, Xa);

  // persistent scan: 2 workgroups x 16 batch rows, ~136 KB dynamic LDS
  size_t smem = (size_t)(16 * HDIM * 3 + 16 * 1024) * sizeof(float)
              + (size_t)(16 * HDIM * 3) * sizeof(_Float16);
  scan_kernel<<<2, 256, smem, stream>>>(x, Xa, waH, waC, wiP, whP, bv, Wf, bf, out);
}
